// FuseModule_41721312313555
// MI455X (gfx1250) — compile-verified, matrix operand order checked by reading, not run on hardware
//
#include <hip/hip_runtime.h>

// ---------------------------------------------------------------------------
// FuseModule (Swin block + channel merging) for gfx1250 / MI455X.
// f16 activations + weights, fp32 accumulation via v_wmma_f32_16x16x32_f16.
// ---------------------------------------------------------------------------

typedef __attribute__((ext_vector_type(16))) _Float16 v16h;
typedef __attribute__((ext_vector_type(8)))  _Float16 v8h;
typedef __attribute__((ext_vector_type(8)))  float    v8f;
typedef __attribute__((ext_vector_type(4)))  int      v4i;

#define WSZ   7
#define NH_   4
#define DIM_  192
#define C_    384
#define B_    32
#define H_    56
#define W_    56
#define T_    49
#define HD_   96
#define HW_   (H_ * W_)        // 3136
#define NTOK  (B_ * HW_)       // 100352
#define NWIN  (B_ * 64)        // 2048

#if defined(__HIP_DEVICE_COMPILE__) && __has_builtin(__builtin_amdgcn_global_load_async_to_lds_b128)
#define HAVE_ASYNC_LDS 1
#endif

// -------------------------- WMMA fragment loaders --------------------------
// 16-bit A fragment 16x32 (ISA 7.12.2): lane l holds row r0+(l&15);
// halfs 0..7  = K[k0 + (l>>4)*8 .. +7], halfs 8..15 = K[k0+16+(l>>4)*8 .. +7].
// B fragment (K x 16) is the mirror with columns across lanes; when B comes
// from a row-major [N][K] weight (X @ W^T) or from K/V^T tiles stored
// K-contiguous, the same loader applies.  Two 16-byte loads per lane.
__device__ __forceinline__ v16h ldfrag(const _Float16* __restrict__ p, int ld,
                                       int r0, int k0, int lane) {
  int r  = r0 + (lane & 15);
  int hh = (lane >> 4) * 8;
  const _Float16* row = p + (size_t)r * ld + k0 + hh;
  v8h lo = *(const v8h*)(row);
  v8h hi = *(const v8h*)(row + 16);
  v16h f;
#pragma unroll
  for (int i = 0; i < 8; ++i) { f[i] = lo[i]; f[8 + i] = hi[i]; }
  return f;
}

// f32 C/D fragment 16x16: lane l holds col n0+(l&15), vgpr r holds row
// m0 + r + 8*(l>>4).  Used to seed accumulators with the attention bias tile.
__device__ __forceinline__ v8f ldfragC(const float* __restrict__ p, int ld,
                                       int m0, int n0, int lane) {
  int n = n0 + (lane & 15);
  int half = lane >> 4;
  v8f c;
#pragma unroll
  for (int r = 0; r < 8; ++r) c[r] = p[(size_t)(m0 + r + 8 * half) * ld + n];
  return c;
}

__device__ __forceinline__ float gelu_f(float x) {
  float x3 = x * x * x;
  return 0.5f * x * (1.0f + tanhf(0.7978845608028654f * (x + 0.044715f * x3)));
}

// window-order row -> token-order row (window reverse)
__device__ __forceinline__ int win2tok(int r) {
  int win = r / T_, t = r - win * T_;
  int b = win >> 6, wrem = win & 63;
  int wh = wrem >> 3, ww = wrem & 7;
  int i = t / WSZ, j = t - i * WSZ;
  return b * HW_ + (wh * WSZ + i) * W_ + (ww * WSZ + j);
}

// ------------------------------ small kernels ------------------------------
__global__ __launch_bounds__(256) void k_cvt(const float* __restrict__ in,
                                             _Float16* __restrict__ out, int n) {
  int i = blockIdx.x * 256 + threadIdx.x;
  if (i < n) out[i] = (_Float16)in[i];
}

// LN1 fused with NCHW->token transpose and window partition.
__global__ __launch_bounds__(128)
void k_ln1(const float* __restrict__ xin, const float* __restrict__ g,
           const float* __restrict__ bb, float* __restrict__ tok,
           _Float16* __restrict__ out) {
  int token = blockIdx.x;
  int b = token / HW_, hw = token - b * HW_;
  int h = hw / W_, w = hw - h * W_;
  int wrow = (b * 64 + (h / WSZ) * 8 + (w / WSZ)) * T_ + (h % WSZ) * WSZ + (w % WSZ);

  const float* base = xin + (size_t)b * C_ * HW_ + hw;
  int c = threadIdx.x;
  float v0 = base[(size_t)(c)       * HW_];
  float v1 = base[(size_t)(c + 128) * HW_];
  float v2 = base[(size_t)(c + 256) * HW_];
  float* trow = tok + (size_t)token * C_;
  trow[c] = v0; trow[c + 128] = v1; trow[c + 256] = v2;

  __shared__ float red[128];
  red[c] = v0 + v1 + v2;
  __syncthreads();
  for (int st = 64; st > 0; st >>= 1) {
    if (c < st) red[c] += red[c + st];
    __syncthreads();
  }
  float mean = red[0] * (1.0f / C_);
  __syncthreads();
  float d0 = v0 - mean, d1 = v1 - mean, d2 = v2 - mean;
  red[c] = d0 * d0 + d1 * d1 + d2 * d2;
  __syncthreads();
  for (int st = 64; st > 0; st >>= 1) {
    if (c < st) red[c] += red[c + st];
    __syncthreads();
  }
  float rstd = rsqrtf(red[0] * (1.0f / C_) + 1e-5f);
  _Float16* q = out + (size_t)wrow * C_;
  q[c]       = (_Float16)(d0 * rstd * g[c]       + bb[c]);
  q[c + 128] = (_Float16)(d1 * rstd * g[c + 128] + bb[c + 128]);
  q[c + 256] = (_Float16)(d2 * rstd * g[c + 256] + bb[c + 256]);
}

// Generic row LayerNorm: fp32 [M][384] -> f16 [M][384].
__global__ __launch_bounds__(128)
void k_ln(const float* __restrict__ in, const float* __restrict__ g,
          const float* __restrict__ bb, _Float16* __restrict__ out) {
  int row = blockIdx.x;
  const float* p = in + (size_t)row * C_;
  int c = threadIdx.x;
  float v0 = p[c], v1 = p[c + 128], v2 = p[c + 256];
  __shared__ float red[128];
  red[c] = v0 + v1 + v2;
  __syncthreads();
  for (int st = 64; st > 0; st >>= 1) {
    if (c < st) red[c] += red[c + st];
    __syncthreads();
  }
  float mean = red[0] * (1.0f / C_);
  __syncthreads();
  float d0 = v0 - mean, d1 = v1 - mean, d2 = v2 - mean;
  red[c] = d0 * d0 + d1 * d1 + d2 * d2;
  __syncthreads();
  for (int st = 64; st > 0; st >>= 1) {
    if (c < st) red[c] += red[c + st];
    __syncthreads();
  }
  float rstd = rsqrtf(red[0] * (1.0f / C_) + 1e-5f);
  _Float16* q = out + (size_t)row * C_;
  q[c]       = (_Float16)(d0 * rstd * g[c]       + bb[c]);
  q[c + 128] = (_Float16)(d1 * rstd * g[c + 128] + bb[c + 128]);
  q[c + 256] = (_Float16)(d2 * rstd * g[c + 256] + bb[c + 256]);
}

// ------------------------------- WMMA GEMM ---------------------------------
// out[M][N] = A[M][K] @ W[N][K]^T  (+ fused epilogue).
// 256 threads = 8 waves; wave w owns m-tile blockIdx.x*8+w and a 64-wide N
// strip (blockIdx.y). A/B fragments loaded straight from global (b128 x2);
// both operands stay L2-resident across re-reads (192 MB L2).
enum { MODE_QKV = 0, MODE_PROJ = 1, MODE_FC1 = 2, MODE_FC2 = 3, MODE_RED = 4 };

template <int MODE>
__global__ __launch_bounds__(256)
void k_gemm(const _Float16* __restrict__ A, const _Float16* __restrict__ W,
            const float* __restrict__ bias, const float* __restrict__ resid,
            float* __restrict__ outf, _Float16* __restrict__ outh,
            int M, int N, int K) {
  int wid = threadIdx.x >> 5, lane = threadIdx.x & 31;
  int tm = blockIdx.x * 8 + wid;
  if (tm * 16 >= M) return;
  int n0 = blockIdx.y * 64;
  const _Float16* Arow = A + (size_t)tm * 16 * K;

  v8f acc[4] = {};
  for (int k0 = 0; k0 < K; k0 += 32) {
    v16h a = ldfrag(Arow, K, 0, k0, lane);
#pragma unroll
    for (int nt = 0; nt < 4; ++nt) {
      v16h b = ldfrag(W, K, n0 + nt * 16, k0, lane);
      acc[nt] = __builtin_amdgcn_wmma_f32_16x16x32_f16(
          false, a, false, b, (short)0, acc[nt], false, false);
    }
  }

  int nl = lane & 15, half = lane >> 4;
#pragma unroll
  for (int nt = 0; nt < 4; ++nt) {
    int n = n0 + nt * 16 + nl;
    float bval;
    if constexpr (MODE == MODE_RED) bval = 0.0f; else bval = bias[n];
#pragma unroll
    for (int r = 0; r < 8; ++r) {
      int m = tm * 16 + r + 8 * half;
      float v = acc[nt][r] + bval;
      if constexpr (MODE == MODE_QKV) {
        outh[(size_t)m * N + n] = (_Float16)v;
      } else if constexpr (MODE == MODE_PROJ) {
        int t = win2tok(m);
        size_t o = (size_t)t * N + n;
        outf[o] = v + resid[o];
      } else if constexpr (MODE == MODE_FC1) {
        outh[(size_t)m * N + n] = (_Float16)gelu_f(v);
      } else if constexpr (MODE == MODE_FC2) {
        size_t o = (size_t)m * N + n;
        outf[o] = v + resid[o];
      } else {  // MODE_RED: token row -> NCHW output
        int b = m / HW_, hw = m - b * HW_;
        outf[((size_t)b * DIM_ + n) * HW_ + hw] = v;
      }
    }
  }
}

// ---------------------------- window attention -----------------------------
// One block = one (window, head). T padded 49->64.
// The relative-position bias (+ key mask) is pre-filled into the score tile
// in LDS and used to SEED the WMMA C accumulators, so the score epilogue is
// branch-free.  K tile is copied global->LDS with async-LDS loads when the
// toolchain exposes the gfx1250 builtin (ASYNCcnt path); Q needs the softmax
// scale and V is stored transposed, so both go through VGPRs.
__global__ __launch_bounds__(128)
void k_attn(const _Float16* __restrict__ qkv, const float* __restrict__ btab,
            const int* __restrict__ ridx, _Float16* __restrict__ o) {
  int win = blockIdx.x >> 2, head = blockIdx.x & 3;
  __shared__ _Float16 Qs[64][HD_];
  __shared__ _Float16 Ks[64][HD_];
  __shared__ _Float16 Vts[HD_][64];       // V transposed: Vts[d][t]
  __shared__ float    Ss[64][65];         // padded: conflict-free rows
  __shared__ _Float16 Ps[64][72];         // ld=72 keeps rows 16B aligned

  int tid = threadIdx.x;
  const float scale = rsqrtf((float)HD_);

#ifdef HAVE_ASYNC_LDS
  // Q (scaled) and V (transposed) via VGPRs; zero K padding rows.
  for (int idx = tid; idx < 64 * HD_; idx += 128) {
    int t = idx / HD_, d = idx - t * HD_;
    _Float16 qv = (_Float16)0.f, vv = (_Float16)0.f;
    if (t < T_) {
      size_t base = (size_t)(win * T_ + t) * (3 * C_) + head * HD_ + d;
      qv = (_Float16)((float)qkv[base] * scale);
      vv = qkv[base + 2 * C_];
    } else {
      Ks[t][d] = (_Float16)0.f;
    }
    Qs[t][d] = qv; Vts[d][t] = vv;
  }
  // K rows 0..48: async global->LDS b128 (ASYNCcnt).
  {
    const _Float16* kbase = qkv + (size_t)win * T_ * (3 * C_) + C_ + head * HD_;
    for (int ch = tid; ch < T_ * (HD_ / 8); ch += 128) {
      int t = ch / (HD_ / 8);
      int d = (ch - t * (HD_ / 8)) * 8;
      __builtin_amdgcn_global_load_async_to_lds_b128(
          (__attribute__((address_space(1))) v4i*)(kbase + (size_t)t * (3 * C_) + d),
          (__attribute__((address_space(3))) v4i*)(&Ks[t][d]), 0, 0);
    }
  }
#else
  for (int idx = tid; idx < 64 * HD_; idx += 128) {
    int t = idx / HD_, d = idx - t * HD_;
    _Float16 qv = (_Float16)0.f, kv = (_Float16)0.f, vv = (_Float16)0.f;
    if (t < T_) {
      size_t base = (size_t)(win * T_ + t) * (3 * C_) + head * HD_ + d;
      qv = (_Float16)((float)qkv[base] * scale);
      kv = qkv[base + C_];
      vv = qkv[base + 2 * C_];
    }
    Qs[t][d] = qv; Ks[t][d] = kv; Vts[d][t] = vv;
  }
#endif

  // Pre-fill score tile with bias / key-mask / zero (becomes WMMA C input).
  for (int idx = tid; idx < 64 * 64; idx += 128) {
    int q = idx >> 6, kk = idx & 63;
    float v;
    if (kk >= T_)      v = -1e30f;
    else if (q >= T_)  v = 0.f;
    else               v = btab[ridx[q * T_ + kk] * NH_ + head];
    Ss[q][kk] = v;
  }

#ifdef HAVE_ASYNC_LDS
#if __has_builtin(__builtin_amdgcn_s_wait_asynccnt)
  __builtin_amdgcn_s_wait_asynccnt(0);
#else
  asm volatile("s_wait_asynccnt 0" ::: "memory");
#endif
#endif
  __syncthreads();

  int wid = tid >> 5, lane = tid & 31;
  int nl = lane & 15, half = lane >> 4;

  // ---- S = (Q*scale) K^T + bias (C seeded from LDS) ----
  {
    v8f acc[4];
#pragma unroll
    for (int nt = 0; nt < 4; ++nt)
      acc[nt] = ldfragC(&Ss[0][0], 65, wid * 16, nt * 16, lane);
    for (int k0 = 0; k0 < HD_; k0 += 32) {
      v16h a = ldfrag(&Qs[0][0], HD_, wid * 16, k0, lane);
#pragma unroll
      for (int nt = 0; nt < 4; ++nt) {
        v16h b = ldfrag(&Ks[0][0], HD_, nt * 16, k0, lane);
        acc[nt] = __builtin_amdgcn_wmma_f32_16x16x32_f16(
            false, a, false, b, (short)0, acc[nt], false, false);
      }
    }
#pragma unroll
    for (int nt = 0; nt < 4; ++nt) {
      int kk = nt * 16 + nl;
#pragma unroll
      for (int r = 0; r < 8; ++r)
        Ss[wid * 16 + r + 8 * half][kk] = acc[nt][r];   // branch-free
    }
  }
  __syncthreads();

  // ---- row softmax (threads 0..63, one row each) ----
  if (tid < 64) {
    float mx = -1e30f;
    for (int kk = 0; kk < 64; ++kk) mx = fmaxf(mx, Ss[tid][kk]);
    float sum = 0.f;
    for (int kk = 0; kk < 64; ++kk) {
      float e = __expf(Ss[tid][kk] - mx);
      sum += e;
      Ss[tid][kk] = e;
    }
    float inv = 1.0f / sum;
    for (int kk = 0; kk < 64; ++kk)
      Ps[tid][kk] = (_Float16)(Ss[tid][kk] * inv);
  }
  __syncthreads();

  // ---- O = P V ----
  {
    v8f acc[6] = {};
    for (int k0 = 0; k0 < 64; k0 += 32) {
      v16h a = ldfrag(&Ps[0][0], 72, wid * 16, k0, lane);
#pragma unroll
      for (int nt = 0; nt < 6; ++nt) {
        v16h b = ldfrag(&Vts[0][0], 64, nt * 16, k0, lane);
        acc[nt] = __builtin_amdgcn_wmma_f32_16x16x32_f16(
            false, a, false, b, (short)0, acc[nt], false, false);
      }
    }
#pragma unroll
    for (int nt = 0; nt < 6; ++nt) {
      int d = nt * 16 + nl;
#pragma unroll
      for (int r = 0; r < 8; ++r) {
        int q = wid * 16 + r + 8 * half;
        if (q < T_)
          o[(size_t)(win * T_ + q) * C_ + head * HD_ + d] = (_Float16)acc[nt][r];
      }
    }
  }
}

// ------------------------------- launcher ----------------------------------
extern "C" void kernel_launch(void* const* d_in, const int* in_sizes, int n_in,
                              void* d_out, int out_size, void* d_ws, size_t ws_size,
                              hipStream_t stream) {
  const float* x        = (const float*)d_in[0];
  const float* norm1_g  = (const float*)d_in[1];
  const float* norm1_b  = (const float*)d_in[2];
  const float* qkv_w    = (const float*)d_in[3];
  const float* qkv_b    = (const float*)d_in[4];
  const float* rbt      = (const float*)d_in[5];
  const float* proj_w   = (const float*)d_in[6];
  const float* proj_b   = (const float*)d_in[7];
  const float* norm2_g  = (const float*)d_in[8];
  const float* norm2_b  = (const float*)d_in[9];
  const float* fc1_w    = (const float*)d_in[10];
  const float* fc1_b    = (const float*)d_in[11];
  const float* fc2_w    = (const float*)d_in[12];
  const float* fc2_b    = (const float*)d_in[13];
  const float* mnorm_g  = (const float*)d_in[14];
  const float* mnorm_b  = (const float*)d_in[15];
  const float* red_w    = (const float*)d_in[16];
  const int*   rel_idx  = (const int*)d_in[17];
  float* out = (float*)d_out;

  // workspace layout (bytes); total ~928.5 MB
  char* ws = (char*)d_ws;
  const size_t SZ_TOK = (size_t)NTOK * C_ * 4;        // 154,140,672
  const size_t SZ_X1  = SZ_TOK;
  const size_t SZ_A   = (size_t)NTOK * C_ * 2;        //  77,070,336
  const size_t SZ_QKV = (size_t)NTOK * 3 * C_ * 2;    // 231,211,008
  const size_t SZ_H   = (size_t)NTOK * 4 * C_ * 2;    // 308,281,344
  float*    tok   = (float*)(ws);
  float*    x1    = (float*)(ws + SZ_TOK);
  _Float16* a16   = (_Float16*)(ws + SZ_TOK + SZ_X1);
  _Float16* qkv16 = (_Float16*)(ws + SZ_TOK + SZ_X1 + SZ_A);
  _Float16* h16   = (_Float16*)(ws + SZ_TOK + SZ_X1 + SZ_A + SZ_QKV);
  _Float16* o16   = h16;  // attention output aliases FC1 buffer (disjoint in time)
  _Float16* wbase = (_Float16*)(ws + SZ_TOK + SZ_X1 + SZ_A + SZ_QKV + SZ_H);
  _Float16* wq  = wbase;
  _Float16* wp  = wq  + (size_t)3 * C_ * C_;
  _Float16* wf1 = wp  + (size_t)C_ * C_;
  _Float16* wf2 = wf1 + (size_t)4 * C_ * C_;
  _Float16* wr  = wf2 + (size_t)C_ * 4 * C_;

  auto cvt = [&](const float* src, _Float16* dst, int n) {
    k_cvt<<<(n + 255) / 256, 256, 0, stream>>>(src, dst, n);
  };
  cvt(qkv_w, wq, 3 * C_ * C_);
  cvt(proj_w, wp, C_ * C_);
  cvt(fc1_w, wf1, 4 * C_ * C_);
  cvt(fc2_w, wf2, C_ * 4 * C_);
  cvt(red_w, wr, DIM_ * C_);

  const int MB = NTOK / 16 / 8;  // 784 m-blocks (8 waves/block)

  // 1) transpose + LN1 + window partition
  k_ln1<<<NTOK, 128, 0, stream>>>(x, norm1_g, norm1_b, tok, a16);
  // 2) QKV GEMM (+bias) -> f16
  k_gemm<MODE_QKV><<<dim3(MB, 18), 256, 0, stream>>>(
      a16, wq, qkv_b, nullptr, nullptr, qkv16, NTOK, 3 * C_, C_);
  // 3) window attention
  k_attn<<<NWIN * NH_, 128, 0, stream>>>(qkv16, rbt, rel_idx, o16);
  // 4) proj GEMM (+bias, +shortcut, window-reverse) -> fp32 x1
  k_gemm<MODE_PROJ><<<dim3(MB, 6), 256, 0, stream>>>(
      o16, wp, proj_b, tok, x1, nullptr, NTOK, C_, C_);
  // 5) LN2
  k_ln<<<NTOK, 128, 0, stream>>>(x1, norm2_g, norm2_b, a16);
  // 6) FC1 (+bias, GELU) -> f16
  k_gemm<MODE_FC1><<<dim3(MB, 24), 256, 0, stream>>>(
      a16, wf1, fc1_b, nullptr, nullptr, h16, NTOK, 4 * C_, C_);
  // 7) FC2 (+bias, +x1) -> fp32 x2 (reuses tok buffer)
  k_gemm<MODE_FC2><<<dim3(MB, 6), 256, 0, stream>>>(
      h16, wf2, fc2_b, x1, tok, nullptr, NTOK, C_, 4 * C_);
  // 8) merge-LN
  k_ln<<<NTOK, 128, 0, stream>>>(tok, mnorm_g, mnorm_b, a16);
  // 9) channel reduction, scatter to NCHW output
  k_gemm<MODE_RED><<<dim3(MB, 3), 256, 0, stream>>>(
      a16, wr, nullptr, nullptr, out, nullptr, NTOK, DIM_, C_);

  (void)in_sizes; (void)n_in; (void)out_size; (void)ws_size;
}
